// RNN_18382460026980
// MI455X (gfx1250) — matrix operand order repaired and checked by reading, not compiled, not run on hardware
//
#include <hip/hip_runtime.h>

// Problem constants (from the reference)
constexpr int kB   = 256;
constexpr int kT   = 2048;
constexpr int kD   = 128;
constexpr int kH   = 64;
constexpr int kLDW = kD + kH; // 192, row stride of W_i2h / W_i2o

typedef __attribute__((ext_vector_type(16))) __bf16 v16bf;
typedef __attribute__((ext_vector_type(16))) float  v16f;
typedef __attribute__((ext_vector_type(8)))  __bf16 v8bf;
typedef __attribute__((ext_vector_type(8)))  float  v8f;
typedef __attribute__((ext_vector_type(4)))  float  v4f;

union F16x  { float f[16]; v4f q[4]; v16f v; };
union Bf16x { v16bf v; uint4 q[2]; unsigned short s[16]; };
union Bf8x  { v8bf v;  uint4 q;    unsigned short s[8];  };

__device__ __forceinline__ float bfbits2f(unsigned short s) {
  union { unsigned u; float f; } o; o.u = ((unsigned)s) << 16; return o.f;
}

// Workspace layout: [0, 24KB) = 24 weight fragments (16 Wx + 8 Wh),
//                   [32KB, ..) = U fragments (bf16), 64 MB.
constexpr size_t kUOffBytes = 32768;

// ---------------------------------------------------------------------------
// Phase 0: one wave converts W_i2h into bf16 B-fragment-native layout.
//   frag f in [0,16): Wx, f = kk*4 + nt   (kk = K/32 chunk of D)
//   frag f in [16,24): Wh, f = 16 + kc*4 + nt
// Per fragment, lane holds col = lane&15, K = 8h+0..7 and 16+8h+0..7 (32 B).
// ---------------------------------------------------------------------------
__global__ __launch_bounds__(32) void rnn_p0_wprep(
    const float* __restrict__ Wih, uint4* __restrict__ WF)
{
  const int lane  = threadIdx.x & 31;
  const int lr    = lane & 15;
  const int koffB = (lane >= 16) ? 8 : 0;

#pragma unroll
  for (int kk = 0; kk < 4; ++kk) {
#pragma unroll
    for (int nt = 0; nt < 4; ++nt) {
      const float* wrow = Wih + (size_t)(nt * 16 + lr) * kLDW + kk * 32 + koffB;
      F16x fb;
      fb.q[0] = *(const v4f*)(wrow + 0);
      fb.q[1] = *(const v4f*)(wrow + 4);
      fb.q[2] = *(const v4f*)(wrow + 16);
      fb.q[3] = *(const v4f*)(wrow + 20);
      Bf16x bp; bp.v = __builtin_convertvector(fb.v, v16bf);
      uint4* dst = WF + ((size_t)(kk * 4 + nt) * 32 + lane) * 2;
      dst[0] = bp.q[0]; dst[1] = bp.q[1];
    }
  }
#pragma unroll
  for (int kc = 0; kc < 2; ++kc) {
#pragma unroll
    for (int nt = 0; nt < 4; ++nt) {
      const float* wrow =
          Wih + (size_t)(nt * 16 + lr) * kLDW + kD + kc * 32 + koffB;
      F16x fb;
      fb.q[0] = *(const v4f*)(wrow + 0);
      fb.q[1] = *(const v4f*)(wrow + 4);
      fb.q[2] = *(const v4f*)(wrow + 16);
      fb.q[3] = *(const v4f*)(wrow + 20);
      Bf16x bp; bp.v = __builtin_convertvector(fb.v, v16bf);
      uint4* dst = WF + ((size_t)(16 + kc * 4 + nt) * 32 + lane) * 2;
      dst[0] = bp.q[0]; dst[1] = bp.q[1];
    }
  }
}

// ---------------------------------------------------------------------------
// Phase 1: U[t][btile][nt] fragments = bias + X[:, t, :] @ Wx^T  (bf16 WMMA)
// One wave per (16-row batch tile, timestep): 16 WMMA, X loaded non-temporal.
// ---------------------------------------------------------------------------
__global__ __launch_bounds__(32) void rnn_p1_xproj(
    const float* __restrict__ X, const float* __restrict__ bih,
    const uint4* __restrict__ WF, uint4* __restrict__ U)
{
  const int btile = blockIdx.x;        // 0..15
  const int t     = blockIdx.y;        // 0..kT-1
  const int lane  = threadIdx.x & 31;
  const int lr    = lane & 15;
  const bool hi   = lane >= 16;
  const int aoff  = hi ? 8 : 0;        // A-matrix K sub-offset per half-wave

  v8f acc[4];
#pragma unroll
  for (int nt = 0; nt < 4; ++nt) {
    const float bv = bih[nt * 16 + lr];
#pragma unroll
    for (int j = 0; j < 8; ++j) acc[nt][j] = bv;
  }

  const float* xrow = X + ((size_t)(btile * 16 + lr) * kT + t) * kD;

#pragma unroll
  for (int kk = 0; kk < 4; ++kk) {
    // A fragment: 16x32 bf16; per-lane two contiguous 8-float groups (NT loads)
    F16x fa;
    fa.q[0] = __builtin_nontemporal_load((const v4f*)(xrow + kk * 32 + aoff));
    fa.q[1] = __builtin_nontemporal_load((const v4f*)(xrow + kk * 32 + aoff + 4));
    fa.q[2] = __builtin_nontemporal_load((const v4f*)(xrow + kk * 32 + aoff + 16));
    fa.q[3] = __builtin_nontemporal_load((const v4f*)(xrow + kk * 32 + aoff + 20));
    Bf16x ap; ap.v = __builtin_convertvector(fa.v, v16bf);

#pragma unroll
    for (int nt = 0; nt < 4; ++nt) {
      const uint4* p = WF + ((size_t)(kk * 4 + nt) * 32 + lane) * 2;
      Bf16x bp; bp.q[0] = p[0]; bp.q[1] = p[1];
      acc[nt] = __builtin_amdgcn_wmma_f32_16x16x32_bf16(
          false, ap.v, false, bp.v, (short)0, acc[nt], false, false);
    }
  }

#pragma unroll
  for (int nt = 0; nt < 4; ++nt) {
    Bf8x up; up.v = __builtin_convertvector(acc[nt], v8bf);
    U[(((size_t)t * 16 + btile) * 4 + nt) * 32 + lane] = up.q;
  }
}

// ---------------------------------------------------------------------------
// Phase 2: per batch tile, sequential scan h_{t+1} = u_t + Wh h_t.
// Wh resident in registers (8 bf16 B-fragments from WF). u_{t+1} loads are
// double-buffered in registers so L2 latency stays off the serial chain.
// h re-swizzled C->A layout through LDS each step. Ends with the projection.
// ---------------------------------------------------------------------------
__global__ __launch_bounds__(32) void rnn_p2_scan(
    const float* __restrict__ X, const float* __restrict__ Wio,
    const float* __restrict__ bio, const uint4* __restrict__ WF,
    const uint4* __restrict__ U, float* __restrict__ out)
{
  __shared__ __attribute__((aligned(16))) unsigned short hsh[16][64];

  const int btile  = blockIdx.x;       // 0..15
  const int lane   = threadIdx.x & 31;
  const int lr     = lane & 15;
  const bool hi    = lane >= 16;
  const int aoff   = hi ? 8 : 0;
  const int rowoff = hi ? 8 : 0;

  // Wh B-fragments (pre-converted by phase 0)
  v16bf bh[2][4];
#pragma unroll
  for (int kc = 0; kc < 2; ++kc) {
#pragma unroll
    for (int nt = 0; nt < 4; ++nt) {
      const uint4* p = WF + ((size_t)(16 + kc * 4 + nt) * 32 + lane) * 2;
      Bf16x bp; bp.q[0] = p[0]; bp.q[1] = p[1];
      bh[kc][nt] = bp.v;
    }
  }

  // h0 = 0 in A-fragment form
  v16bf ha[2];
  {
    Bf16x z;
#pragma unroll
    for (int i = 0; i < 16; ++i) z.s[i] = 0;
    ha[0] = z.v; ha[1] = z.v;
  }

  // Preload u_0 fragments (register double buffer)
  uint4 ub[4];
  {
    const size_t base0 = (size_t)btile * 4 * 32 + lane;
#pragma unroll
    for (int nt = 0; nt < 4; ++nt) ub[nt] = U[base0 + (size_t)nt * 32];
  }

  for (int t = 0; t < kT - 1; ++t) {
    // Issue next step's u loads now; they complete under this step's compute.
    const int tn = (t + 1 < kT - 1) ? (t + 1) : t;
    const size_t nbase = (((size_t)tn * 16 + btile) * 4) * 32 + lane;
    uint4 un[4];
#pragma unroll
    for (int nt = 0; nt < 4; ++nt) un[nt] = U[nbase + (size_t)nt * 32];
    if (t + 8 < kT - 1)  // long-range: keep the U stream warm near the WGP
      __builtin_prefetch(&U[(((size_t)(t + 8) * 16 + btile) * 4) * 32 + lane], 0, 1);

    // acc = u_t (bf16 fragments -> fp32 accumulators)
    v8f acc[4];
#pragma unroll
    for (int nt = 0; nt < 4; ++nt) {
      Bf8x up; up.q = ub[nt];
      acc[nt] = __builtin_convertvector(up.v, v8f);
    }

    // acc += h_t @ Wh^T
#pragma unroll
    for (int nt = 0; nt < 4; ++nt) {
      acc[nt] = __builtin_amdgcn_wmma_f32_16x16x32_bf16(
          false, ha[0], false, bh[0][nt], (short)0, acc[nt], false, false);
      acc[nt] = __builtin_amdgcn_wmma_f32_16x16x32_bf16(
          false, ha[1], false, bh[1][nt], (short)0, acc[nt], false, false);
    }

    // C-layout -> LDS (packed cvt, then 2-byte scatter within the tile)
#pragma unroll
    for (int nt = 0; nt < 4; ++nt) {
      Bf8x hp; hp.v = __builtin_convertvector(acc[nt], v8bf);
#pragma unroll
      for (int j = 0; j < 8; ++j)
        hsh[j + rowoff][nt * 16 + lr] = hp.s[j];
    }
    __syncthreads();

    // LDS -> A-layout fragments for next step (2x ds_load_b128 per chunk)
#pragma unroll
    for (int kc = 0; kc < 2; ++kc) {
      Bf16x hp;
      hp.q[0] = *(const uint4*)&hsh[lr][kc * 32 + aoff];
      hp.q[1] = *(const uint4*)&hsh[lr][kc * 32 + aoff + 16];
      ha[kc] = hp.v;
    }
    __syncthreads();

    // Rotate the u double buffer
#pragma unroll
    for (int nt = 0; nt < 4; ++nt) ub[nt] = un[nt];
  }

  // Output projection: out[b] = [x_{T-1}, h] . Wio[0,:] + bio[0]
  if (lane < 16) {
    const int b = btile * 16 + lane;
    const float* xrow = X + ((size_t)b * kT + (kT - 1)) * kD;
    float acc = bio[0];
#pragma unroll 4
    for (int k = 0; k < kD; ++k) acc += xrow[k] * Wio[k];
#pragma unroll 4
    for (int k = 0; k < kH; ++k) acc += bfbits2f(hsh[lane][k]) * Wio[kD + k];
    out[b] = acc;
  }
}

extern "C" void kernel_launch(void* const* d_in, const int* in_sizes, int n_in,
                              void* d_out, int out_size, void* d_ws, size_t ws_size,
                              hipStream_t stream) {
  const float* X   = (const float*)d_in[0];  // [B, T, D]
  const float* Wih = (const float*)d_in[1];  // [H, D+H]
  const float* bih = (const float*)d_in[2];  // [H]
  const float* Wio = (const float*)d_in[3];  // [C, D+H]
  const float* bio = (const float*)d_in[4];  // [C]
  float* out = (float*)d_out;                // [B]

  uint4* WF = (uint4*)d_ws;                              // 24 KB of fragments
  uint4* U  = (uint4*)((char*)d_ws + kUOffBytes);        // 64 MB bf16 fragments

  rnn_p0_wprep<<<1, 32, 0, stream>>>(Wih, WF);
  dim3 g1(kB / 16, kT);
  rnn_p1_xproj<<<g1, 32, 0, stream>>>(X, bih, WF, U);
  rnn_p2_scan<<<dim3(kB / 16), 32, 0, stream>>>(X, Wio, bio, WF, U, out);
}
